// GNN_BioGPT2_51917564674532
// MI455X (gfx1250) — matrix operand order repaired and checked by reading, not compiled, not run on hardware
//
#include <hip/hip_runtime.h>
#include <hip/hip_bf16.h>

// ---------------------------------------------------------------------------
// Types for CDNA5 WMMA (wave32): v_wmma_f32_16x16x32_bf16
// ---------------------------------------------------------------------------
typedef __attribute__((ext_vector_type(16))) __bf16 v16bf;
typedef __attribute__((ext_vector_type(8)))  float  v8f;

union Frag { v16bf v; unsigned int u[8]; };
union Acc  { v8f   v; float        f[8]; };

__device__ __forceinline__ unsigned short f2bf(float x) {
    unsigned int b = __float_as_uint(x);
    b += 0x7FFFu + ((b >> 16) & 1u);      // round-to-nearest-even
    return (unsigned short)(b >> 16);
}

// LDS offset of a generic pointer that points into LDS (generic -> AS(3) cast
// strips the aperture, leaving the 32-bit LDS byte offset).
__device__ __forceinline__ unsigned lds_offset_of(void* p) {
    return (unsigned)(unsigned long long)(__attribute__((address_space(3))) void*)p;
}

// CDNA5 async copy: global -> LDS, 16 bytes per lane, tracked by ASYNCcnt.
__device__ __forceinline__ void async_b128(void* lds_dst, const void* gsrc) {
    const unsigned      dst = lds_offset_of(lds_dst);
    const unsigned long long src = (unsigned long long)gsrc;
    asm volatile("global_load_async_to_lds_b128 %0, %1, off"
                 :: "v"(dst), "v"(src) : "memory");
}
__device__ __forceinline__ void wait_async0() {
    asm volatile("s_wait_asynccnt 0x0" ::: "memory");
}

// ---------------------------------------------------------------------------
// bf16 WMMA GEMM:  C[M,N] = A[M,K] * Bt[N,K]^T   (A, Bt already bf16)
//   block tile 128(M) x 64(N), K-step 32, 8 waves arranged 4(m) x 2(n),
//   each wave computes a 32x32 output (4 accumulators, 2x fragment reuse).
//   Tiles staged with double-buffered global_load_async_to_lds_b128.
// Requires M%128==0, K%32==0, N%64==0.
// ---------------------------------------------------------------------------
#define LDPB 80                       // LDS row pitch in bytes (64B data + pad)
#define ABYTES (128 * LDPB)           // 10240
#define BUFBYTES (ABYTES + 64 * LDPB) // 15360 per buffer

__device__ __forceinline__ void stage_tile(char* buf, const char* Ab, const char* Bb,
                                           long m0, int n0, long Kb, int kk, int t)
{
    // A tile: 128 rows x 64 bytes  -> 512 x b128 chunks, 2 per thread
#pragma unroll
    for (int q = 0; q < 2; ++q) {
        const int c = t + q * 256;
        const int row = c >> 2, o = (c & 3) * 16;
        async_b128(buf + row * LDPB + o,
                   Ab + (m0 + row) * Kb + (long)kk * 2 + o);
    }
    // B tile: 64 rows x 64 bytes -> 256 x b128 chunks, 1 per thread
    {
        const int n = t >> 2, o = (t & 3) * 16;
        async_b128(buf + ABYTES + n * LDPB + o,
                   Bb + (n0 + n) * Kb + (long)kk * 2 + o);
    }
}

__global__ __launch_bounds__(256)
void gemm_bf16(const unsigned short* __restrict__ A, const unsigned short* __restrict__ Bt,
               float* __restrict__ C, int K, int Nn)
{
    __shared__ __align__(16) char sm[2 * BUFBYTES];

    const int t    = threadIdx.x;
    const int lane = t & 31;
    const int w    = t >> 5;
    const int mW   = w & 3;                 // 4 m-waves: rows mW*32..+31
    const int nW   = w >> 2;                // 2 n-waves: cols nW*32..+31
    const long m0  = (long)blockIdx.y * 128;
    const int  n0  = blockIdx.x * 64;
    const char* Ab = (const char*)A;
    const char* Bb = (const char*)Bt;
    const long Kb  = (long)K * 2;           // bytes per row

    Acc acc[2][2];
#pragma unroll
    for (int s = 0; s < 2; ++s)
#pragma unroll
        for (int u = 0; u < 2; ++u)
#pragma unroll
            for (int i = 0; i < 8; ++i) acc[s][u].f[i] = 0.f;

    const int nk = K / 32;
    stage_tile(sm, Ab, Bb, m0, n0, Kb, 0, t);           // prefetch tile 0

    for (int i = 0; i < nk; ++i) {
        wait_async0();                                   // my async writes done
        __syncthreads();                                 // tile i visible to all
        if (i + 1 < nk)                                  // stream next tile
            stage_tile(sm + ((i + 1) & 1) * BUFBYTES, Ab, Bb, m0, n0, Kb, (i + 1) * 32, t);

        const char* base = sm + (i & 1) * BUFBYTES;
        const int kaOff = (lane >= 16) ? 16 : 0;         // A: +8 halves
        const int kbOff = (lane >= 16) ? 16 : 0;         // B: +8 halves

        Frag fa[2], fb[2];
#pragma unroll
        for (int s = 0; s < 2; ++s) {                    // A frags (2 m-subtiles)
            const char* p = base + (mW * 32 + s * 16 + (lane & 15)) * LDPB + kaOff;
            const uint4 r0 = *(const uint4*)p;
            const uint4 r1 = *(const uint4*)(p + 32);
            fa[s].u[0] = r0.x; fa[s].u[1] = r0.y; fa[s].u[2] = r0.z; fa[s].u[3] = r0.w;
            fa[s].u[4] = r1.x; fa[s].u[5] = r1.y; fa[s].u[6] = r1.z; fa[s].u[7] = r1.w;
        }
#pragma unroll
        for (int u = 0; u < 2; ++u) {                    // B frags (2 n-subtiles)
            const char* p = base + ABYTES + (nW * 32 + u * 16 + (lane & 15)) * LDPB + kbOff;
            const uint4 r0 = *(const uint4*)p;
            const uint4 r1 = *(const uint4*)(p + 32);
            fb[u].u[0] = r0.x; fb[u].u[1] = r0.y; fb[u].u[2] = r0.z; fb[u].u[3] = r0.w;
            fb[u].u[4] = r1.x; fb[u].u[5] = r1.y; fb[u].u[6] = r1.z; fb[u].u[7] = r1.w;
        }
#pragma unroll
        for (int s = 0; s < 2; ++s)
#pragma unroll
            for (int u = 0; u < 2; ++u)
                acc[s][u].v = __builtin_amdgcn_wmma_f32_16x16x32_bf16(
                    false, fa[s].v, false, fb[u].v, (short)0, acc[s][u].v, false, false);
        // No trailing barrier needed: next overwrite of this buffer is issued
        // only after the *next* iteration's __syncthreads().
    }

    // store 32x32 per wave per the 16x16 f32 C/D layout
#pragma unroll
    for (int s = 0; s < 2; ++s) {
        const long rbase = m0 + mW * 32 + s * 16 + ((lane >= 16) ? 8 : 0);
#pragma unroll
        for (int u = 0; u < 2; ++u) {
            const int cn = n0 + nW * 32 + u * 16 + (lane & 15);
#pragma unroll
            for (int r = 0; r < 8; ++r)
                C[(rbase + r) * Nn + cn] = acc[s][u].f[r];
        }
    }
}

// ---------------------------------------------------------------------------
// fp32 -> bf16 conversion (x): 4 elements per thread
// ---------------------------------------------------------------------------
__global__ void cvt_bf16(const float* __restrict__ in, unsigned short* __restrict__ out, long n)
{
    const long i = ((long)blockIdx.x * blockDim.x + threadIdx.x) * 4;
    if (i + 3 >= n) return;
    const float4 v = *(const float4*)(in + i);
    uint2 p;
    p.x = (unsigned)f2bf(v.x) | ((unsigned)f2bf(v.y) << 16);
    p.y = (unsigned)f2bf(v.z) | ((unsigned)f2bf(v.w) << 16);
    *(uint2*)(out + i) = p;
}

// fp32 [K,N] -> bf16 transposed [N,K]
__global__ void cvt_bf16_T(const float* __restrict__ in, unsigned short* __restrict__ out,
                           int K, int N)
{
    const long idx = (long)blockIdx.x * blockDim.x + threadIdx.x;
    if (idx >= (long)K * N) return;
    const int k = (int)(idx % K), n = (int)(idx / K);
    out[(long)n * K + k] = f2bf(in[(long)k * N + n]);
}

// ---------------------------------------------------------------------------
// Per-graph degree (incoming + self loop) -> D^-1/2
// ---------------------------------------------------------------------------
__global__ void deg_dinv(const long long* __restrict__ ei, float* __restrict__ dinv)
{
    __shared__ int cnt[100];
    const int g = blockIdx.x, t = threadIdx.x;
    for (int n = t; n < 100; n += blockDim.x) cnt[n] = 0;
    __syncthreads();
    const long long* dst = ei + (long)g * 3200 + 1600;
    for (int e = t; e < 1600; e += blockDim.x)
        atomicAdd(&cnt[(int)dst[e]], 1);
    __syncthreads();
    for (int n = t; n < 100; n += blockDim.x) {
        const float d = (float)(cnt[n] + 1);
        dinv[g * 100 + n] = rsqrtf(fmaxf(d, 1e-12f));
    }
}

// ---------------------------------------------------------------------------
// GCN aggregation -> relu -> bf16 (feeds the second WMMA GEMM directly)
// ---------------------------------------------------------------------------
__global__ __launch_bounds__(256)
void gcn_agg(const float* __restrict__ XW, const long long* __restrict__ ei,
             const float* __restrict__ dinv, const float* __restrict__ b,
             unsigned short* __restrict__ H)
{
    const int gn = blockIdx.x;           // 0..25599
    const int g = gn / 100, n = gn % 100;
    const int t = threadIdx.x;
    const long long* es = ei + (long)g * 3200;
    const long long* ed = es + 1600;
    const float* dv   = dinv + g * 100;
    const float  dn   = dv[n];
    const float* base = XW + (long)g * 100 * 512;

    float a0 = base[(long)n * 512 + t]       * dn * dn;
    float a1 = base[(long)n * 512 + t + 256] * dn * dn;
    for (int e = 0; e < 1600; ++e) {
        if ((int)ed[e] == n) {
            const int s = (int)es[e];
            const float wgt = dv[s] * dn;
            a0 += base[(long)s * 512 + t]       * wgt;
            a1 += base[(long)s * 512 + t + 256] * wgt;
        }
    }
    const long o = (long)gn * 512;
    H[o + t]       = f2bf(fmaxf(a0 + b[t], 0.f));
    H[o + t + 256] = f2bf(fmaxf(a1 + b[t + 256], 0.f));
}

// ---------------------------------------------------------------------------
// asrc/adst logits: per-row dot with a_src / a_dst (D=256)
// ---------------------------------------------------------------------------
__global__ void row_dots(const float* __restrict__ h2, const float* __restrict__ a_src,
                         const float* __restrict__ a_dst,
                         float* __restrict__ asr, float* __restrict__ ads)
{
    const int r = blockIdx.x * blockDim.x + threadIdx.x;
    if (r >= 25600) return;
    const float* row = h2 + (long)r * 256;
    float s0 = 0.f, s1 = 0.f;
    for (int k = 0; k < 256; ++k) {
        const float v = row[k];
        s0 += v * a_src[k];
        s1 += v * a_dst[k];
    }
    asr[r] = s0; ads[r] = s1;
}

// ---------------------------------------------------------------------------
// GAT attention: per-graph softmax over incoming edges (1600 + 100 self loops)
// plus patch scores = segment-mean of alpha over both endpoints.
// ---------------------------------------------------------------------------
__global__ __launch_bounds__(256)
void gat_att(const long long* __restrict__ ei, const float* __restrict__ asr,
             const float* __restrict__ ads, float* __restrict__ alpha,
             float* __restrict__ score, float* __restrict__ score2)
{
    __shared__ float asL[100], adL[100], mL[100], sL[100];
    __shared__ float valL[1700], alL[1700];
    __shared__ short srcL[1700], dstL[1700];

    const int g = blockIdx.x, t = threadIdx.x;
    const long long* es = ei + (long)g * 3200;
    const long long* ed = es + 1600;

    for (int n = t; n < 100; n += 256) { asL[n] = asr[g * 100 + n]; adL[n] = ads[g * 100 + n]; }
    for (int e = t; e < 1700; e += 256) {
        int s, d;
        if (e < 1600) { s = (int)es[e]; d = (int)ed[e]; }
        else          { s = d = e - 1600; }
        srcL[e] = (short)s; dstL[e] = (short)d;
    }
    __syncthreads();
    for (int e = t; e < 1700; e += 256) {
        const float v = asL[srcL[e]] + adL[dstL[e]];
        valL[e] = (v > 0.f) ? v : 0.2f * v;            // leaky_relu(0.2)
    }
    __syncthreads();
    if (t < 100) {                                     // segment max then sum
        float m = -3.4e38f;
        for (int e = 0; e < 1700; ++e) if (dstL[e] == t) m = fmaxf(m, valL[e]);
        mL[t] = m;
        float s = 0.f;
        for (int e = 0; e < 1700; ++e) if (dstL[e] == t) s += __expf(valL[e] - m);
        sL[t] = s;
    }
    __syncthreads();
    for (int e = t; e < 1700; e += 256) {
        const float a = __expf(valL[e] - mL[dstL[e]]) / sL[dstL[e]];
        alL[e] = a;
        alpha[(long)g * 1700 + e] = a;
    }
    __syncthreads();
    if (t < 100) {                                     // patch score (segment mean)
        float att = 0.f, cnt = 0.f;
        for (int e = 0; e < 1700; ++e) {
            if (srcL[e] == t) { att += alL[e]; cnt += 1.f; }
            if (dstL[e] == t) { att += alL[e]; cnt += 1.f; }
        }
        const float sc = att / cnt;
        score[g * 100 + t] = sc;
        score2[g * 100 + t] = sc;
    }
}

// ---------------------------------------------------------------------------
// GAT aggregation: out[n] = sum_{e:dst=n} alpha[e]*h2[src] + alpha_self*h2[n] + b
// ---------------------------------------------------------------------------
__global__ __launch_bounds__(256)
void gat_agg(const float* __restrict__ h2, const long long* __restrict__ ei,
             const float* __restrict__ alpha, const float* __restrict__ b,
             float* __restrict__ out)
{
    const int gn = blockIdx.x;
    const int g = gn / 100, n = gn % 100;
    const int t = threadIdx.x;                         // feature 0..255
    const long long* es = ei + (long)g * 3200;
    const long long* ed = es + 1600;
    const float* al   = alpha + (long)g * 1700;
    const float* base = h2 + (long)g * 100 * 256;

    float acc = al[1600 + n] * base[(long)n * 256 + t];
    for (int e = 0; e < 1600; ++e) {
        if ((int)ed[e] == n)
            acc += al[e] * base[(long)((int)es[e]) * 256 + t];
    }
    out[(long)gn * 256 + t] = acc + b[t];
}

// ---------------------------------------------------------------------------
// Top-50 of 25600 scores (single block, 50 rounds of argmax)
// ---------------------------------------------------------------------------
__global__ __launch_bounds__(256)
void topk50(float* __restrict__ sc, int* __restrict__ idx)
{
    __shared__ float bm[256];
    __shared__ int   bi[256];
    const int t = threadIdx.x;
    for (int it = 0; it < 50; ++it) {
        float m = -3.4e38f; int mi = 0;
        for (int i = t; i < 25600; i += 256) {
            const float v = sc[i];
            if (v > m) { m = v; mi = i; }
        }
        bm[t] = m; bi[t] = mi;
        __syncthreads();
        for (int off = 128; off > 0; off >>= 1) {
            if (t < off && bm[t + off] > bm[t]) { bm[t] = bm[t + off]; bi[t] = bi[t + off]; }
            __syncthreads();
        }
        if (t == 0) { idx[it] = bi[0]; sc[bi[0]] = -3.4e38f; }
        __syncthreads();
    }
}

// ---------------------------------------------------------------------------
// Project only the 50 selected rows: out[r] = gnn[idx[r]] @ W_proj + b_proj
// ---------------------------------------------------------------------------
__global__ __launch_bounds__(256)
void proj_gather(const float* __restrict__ gnn, const int* __restrict__ idx,
                 const float* __restrict__ W, const float* __restrict__ b,
                 float* __restrict__ out)
{
    const int r = blockIdx.x;                          // 0..49
    const int c = blockIdx.y * 256 + threadIdx.x;      // 0..1023
    const float* row = gnn + (long)idx[r] * 256;
    float acc = b[c];
    for (int k = 0; k < 256; ++k)
        acc += row[k] * W[(long)k * 1024 + c];
    out[(long)r * 1024 + c] = acc;
}

// ---------------------------------------------------------------------------
// Launcher
// ---------------------------------------------------------------------------
extern "C" void kernel_launch(void* const* d_in, const int* in_sizes, int n_in,
                              void* d_out, int out_size, void* d_ws, size_t ws_size,
                              hipStream_t stream)
{
    const float*     x      = (const float*)d_in[0];      // [256,100,2048]
    const long long* ei     = (const long long*)d_in[1];  // [256,2,1600] int64
    const float*     W_gcn  = (const float*)d_in[2];      // [2048,512]
    const float*     b_gcn  = (const float*)d_in[3];      // [512]
    const float*     W_gat  = (const float*)d_in[4];      // [512,256]
    const float*     b_gat  = (const float*)d_in[5];      // [256]
    const float*     a_src  = (const float*)d_in[6];      // [256]
    const float*     a_dst  = (const float*)d_in[7];      // [256]
    const float*     W_proj = (const float*)d_in[8];      // [256,1024]
    const float*     b_proj = (const float*)d_in[9];      // [1024]
    float*           out    = (float*)d_out;              // [50,1024]

    char* ws = (char*)d_ws;
    size_t off = 0;
    auto take = [&](size_t bytes) { char* p = ws + off; off += (bytes + 255) & ~(size_t)255; return p; };

    unsigned short* xb     = (unsigned short*)take(25600ull * 2048 * 2); // x in bf16 (L2-resident)
    unsigned short* WgcnT  = (unsigned short*)take(512ull * 2048 * 2);   // W_gcn^T bf16 [512][2048]
    unsigned short* WgatT  = (unsigned short*)take(256ull * 512 * 2);    // W_gat^T bf16 [256][512]
    float*          XW     = (float*)take(25600ull * 512 * 4);           // x @ W_gcn
    unsigned short* Hb     = (unsigned short*)take(25600ull * 512 * 2);  // GCN out (relu, bf16)
    float*          h2     = (float*)take(25600ull * 256 * 4);           // h @ W_gat
    float*          gnn    = (float*)take(25600ull * 256 * 4);           // GAT out
    float*          dinv   = (float*)take(25600ull * 4);
    float*          asr    = (float*)take(25600ull * 4);
    float*          ads    = (float*)take(25600ull * 4);
    float*          alpha  = (float*)take(256ull * 1700 * 4);
    float*          score  = (float*)take(25600ull * 4);
    float*          score2 = (float*)take(25600ull * 4);
    int*            tkidx  = (int*)take(64 * 4);
    (void)ws_size; (void)in_sizes; (void)n_in; (void)out_size; (void)score;

    // 0) precision/layout prep: x -> bf16; weights -> bf16 transposed [N][K]
    cvt_bf16  <<<(25600ull * 2048 / 4 + 255) / 256, 256, 0, stream>>>(x, xb, 25600ll * 2048);
    cvt_bf16_T<<<(2048 * 512 + 255) / 256,          256, 0, stream>>>(W_gcn, WgcnT, 2048, 512);
    cvt_bf16_T<<<(512 * 256 + 255) / 256,           256, 0, stream>>>(W_gat, WgatT, 512, 256);

    // 1) degree norms
    deg_dinv<<<256, 128, 0, stream>>>(ei, dinv);
    // 2) GCN GEMM: [25600,2048] x [2048,512]  (bf16 WMMA, async-LDS pipeline)
    gemm_bf16<<<dim3(512 / 64, 25600 / 128), 256, 0, stream>>>(xb, WgcnT, XW, 2048, 512);
    // 3) GCN normalize/aggregate + bias + relu -> bf16
    gcn_agg<<<25600, 256, 0, stream>>>(XW, ei, dinv, b_gcn, Hb);
    // 4) GAT GEMM: [25600,512] x [512,256]  (bf16 WMMA, async-LDS pipeline)
    gemm_bf16<<<dim3(256 / 64, 25600 / 128), 256, 0, stream>>>(Hb, WgatT, h2, 512, 256);
    // 5) attention logits
    row_dots<<<100, 256, 0, stream>>>(h2, a_src, a_dst, asr, ads);
    // 6) segment softmax + patch scores
    gat_att<<<256, 256, 0, stream>>>(ei, asr, ads, alpha, score, score2);
    // 7) GAT aggregation
    gat_agg<<<25600, 256, 0, stream>>>(h2, ei, alpha, b_gat, gnn);
    // 8) top-50 scores (before projection — saves the full [25600,1024] GEMM)
    topk50<<<1, 256, 0, stream>>>(score2, tkidx);
    // 9) project only selected rows
    proj_gather<<<dim3(50, 4), 256, 0, stream>>>(gnn, tkidx, W_proj, b_proj, out);
}
